// EncoderGNN_63797444214959
// MI455X (gfx1250) — compile-verified, matrix operand order checked by reading, not run on hardware
//
#include <hip/hip_runtime.h>
#include <hip/hip_bf16.h>
#include <math.h>

// ---- sizes ----
#define Bx 2
#define Nn 128
#define Cc 256
#define NODE_IN 128
#define EDGE_IN 64
#define Ll 3
#define KN 101
#define KE 7
#define ALPHA 0.2f
#define EPSl 1e-5f
#define Mrows (Bx * Nn * Nn)   // 32768

typedef _Float16 v16h __attribute__((ext_vector_type(16)));
typedef _Float16 v8h  __attribute__((ext_vector_type(8)));
typedef float    v8f  __attribute__((ext_vector_type(8)));

__device__ __forceinline__ float lrelu(float x) { return x > 0.f ? x : ALPHA * x; }

// Load a 16x32 (or 32x16) 16-bit WMMA fragment for this lane directly from a
// packed f16 row. CDNA5 wave32 layout: lanes 0-15 K-base 0, lanes 16-31 K-base 8;
// element e: K = kb + (e>>3)*16 + (e&7)  -> two contiguous 16-byte chunks.
__device__ __forceinline__ v16h load_frag(const _Float16* rowk, int lane) {
    const v8h* p = reinterpret_cast<const v8h*>(rowk + (((lane >> 4)) << 3));
    v8h lo = p[0];     // K = kb+0..7
    v8h hi = p[2];     // K = kb+16..23
    return __builtin_shufflevector(lo, hi, 0, 1, 2, 3, 4, 5, 6, 7,
                                   8, 9, 10, 11, 12, 13, 14, 15);
}

// =====================================================================
// conversions
// =====================================================================
__global__ void k_f32_to_f16(const float* __restrict__ in, _Float16* __restrict__ out,
                             int n) {
    int i = blockIdx.x * blockDim.x + threadIdx.x;
    if (i < n) out[i] = (_Float16)in[i];
}

// strided weight slice (rows x cols, leading dim ldin) -> packed f16 (rows x cols)
__global__ void k_w_to_f16(const float* __restrict__ in, int ldin, int cols,
                           _Float16* __restrict__ out) {
    int r = blockIdx.x;
    for (int c = threadIdx.x; c < cols; c += blockDim.x)
        out[r * cols + c] = (_Float16)in[(size_t)r * ldin + c];
}

// =====================================================================
// 1) node projection: nodes0 = nodes @ pnW^T + pnb   (256 x 256, K=128)
// =====================================================================
__global__ void k_node_proj(const float* __restrict__ nodes_in,
                            const float* __restrict__ pnW,
                            const float* __restrict__ pnb,
                            float* __restrict__ out) {
    __shared__ float sn[NODE_IN];
    int row = blockIdx.x;
    int tid = threadIdx.x;
    if (tid < NODE_IN) sn[tid] = nodes_in[row * NODE_IN + tid];
    __syncthreads();
    const float* w = pnW + tid * NODE_IN;
    float acc = pnb[tid];
    #pragma unroll 8
    for (int k = 0; k < NODE_IN; ++k) acc += sn[k] * w[k];
    out[row * Cc + tid] = acc;
}

// =====================================================================
// 2) WMMA edge projection: E = edges_raw_h(32768 x 64) @ peW_h^T + peb
//    Writes fp32 result + f16 shadow for the next GEMM's A operand.
// =====================================================================
__global__ void k_edge_proj_wmma(const _Float16* __restrict__ Xh,  // (M,64) f16
                                 const _Float16* __restrict__ Wh,  // (256,64) f16
                                 const float* __restrict__ bias,
                                 float* __restrict__ Y,            // (M,256)
                                 _Float16* __restrict__ Yh) {      // (M,256)
    int tid  = threadIdx.x;
    int wv   = tid >> 5, lane = tid & 31;
    int rb   = wv >> 2, cb = wv & 3;
    int rowBase = blockIdx.x * 32 + rb * 16;
    const _Float16* arow = Xh + (size_t)(rowBase + (lane & 15)) * EDGE_IN;

    v8f acc[4] = {};
    #pragma unroll
    for (int kk = 0; kk < EDGE_IN; kk += 32) {
        v16h a = load_frag(arow + kk, lane);
        #pragma unroll
        for (int ct = 0; ct < 4; ++ct) {
            int col0 = cb * 64 + ct * 16;
            v16h bf = load_frag(Wh + (size_t)(col0 + (lane & 15)) * EDGE_IN + kk, lane);
            acc[ct] = __builtin_amdgcn_wmma_f32_16x16x32_f16(
                false, a, false, bf, (short)0, acc[ct], false, false);
        }
    }
    #pragma unroll
    for (int ct = 0; ct < 4; ++ct) {
        int col = cb * 64 + ct * 16 + (lane & 15);
        float bcol = bias[col];
        #pragma unroll
        for (int r = 0; r < 8; ++r) {
            int row = rowBase + r + ((lane >> 4) << 3);
            float v = acc[ct][r] + bcol;
            Y[(size_t)row * Cc + col]  = v;
            Yh[(size_t)row * Cc + col] = (_Float16)v;
        }
    }
}

// =====================================================================
// 3) fused layer edge update (dominant GEMM, f16 in / fp32 acc):
//    ne  = lrelu(ai_i + E @ Wb2^T + aj_j) * mask
//    out = LayerNorm(ne + E) over C  (LN fused via LDS ds_add_f32)
//    also emits f16 shadow of out for the next layer's A operand.
// =====================================================================
__global__ void k_edge_layer_wmma(const float* __restrict__ X,      // E fp32 (M,256)
                                  const _Float16* __restrict__ Xh,  // E f16  (M,256)
                                  const _Float16* __restrict__ Wh,  // Wb2 f16 (256,256)
                                  const float* __restrict__ ai,
                                  const float* __restrict__ aj,
                                  const float* __restrict__ mask,
                                  const float* __restrict__ g,
                                  const float* __restrict__ bvec,
                                  float* __restrict__ Y,
                                  _Float16* __restrict__ Yh) {
    __shared__ float s_sum[32];
    __shared__ float s_sq[32];
    int tid = threadIdx.x;
    int wv = tid >> 5, lane = tid & 31;
    int rb = wv >> 2, cb = wv & 3;
    int rowBase = blockIdx.x * 32 + rb * 16;
    const _Float16* arow = Xh + (size_t)(rowBase + (lane & 15)) * Cc;

    if (tid < 32) { s_sum[tid] = 0.f; s_sq[tid] = 0.f; }

    v8f acc[4] = {};
    #pragma unroll
    for (int kk = 0; kk < Cc; kk += 32) {
        v16h a = load_frag(arow + kk, lane);
        #pragma unroll
        for (int ct = 0; ct < 4; ++ct) {
            int col0 = cb * 64 + ct * 16;
            v16h bf = load_frag(Wh + (size_t)(col0 + (lane & 15)) * Cc + kk, lane);
            acc[ct] = __builtin_amdgcn_wmma_f32_16x16x32_f16(
                false, a, false, bf, (short)0, acc[ct], false, false);
        }
    }
    __syncthreads();

    // epilogue: bias terms + lrelu + mask + residual, stash into acc
    #pragma unroll
    for (int ct = 0; ct < 4; ++ct) {
        int col = cb * 64 + ct * 16 + (lane & 15);
        #pragma unroll
        for (int r = 0; r < 8; ++r) {
            int rowL = rb * 16 + r + ((lane >> 4) << 3);
            int m = blockIdx.x * 32 + rowL;
            int bb = m >> 14, i = (m >> 7) & 127, j = m & 127;
            float x = acc[ct][r]
                    + ai[((bb << 7) + i) * Cc + col]
                    + aj[((bb << 7) + j) * Cc + col];
            x = lrelu(x);
            x *= mask[((bb << 7) + i) * Nn + j];
            x += X[(size_t)m * Cc + col];
            acc[ct][r] = x;
        }
    }
    // per-row sum / sumsq over C=256 via LDS float atomics (ds_add_f32)
    #pragma unroll
    for (int r = 0; r < 8; ++r) {
        int rowL = rb * 16 + r + ((lane >> 4) << 3);
        float ls = acc[0][r] + acc[1][r] + acc[2][r] + acc[3][r];
        float lq = acc[0][r] * acc[0][r] + acc[1][r] * acc[1][r]
                 + acc[2][r] * acc[2][r] + acc[3][r] * acc[3][r];
        atomicAdd(&s_sum[rowL], ls);
        atomicAdd(&s_sq[rowL], lq);
    }
    __syncthreads();

    #pragma unroll
    for (int ct = 0; ct < 4; ++ct) {
        int col = cb * 64 + ct * 16 + (lane & 15);
        float gg = g[col], bb2 = bvec[col];
        #pragma unroll
        for (int r = 0; r < 8; ++r) {
            int rowL = rb * 16 + r + ((lane >> 4) << 3);
            int m = blockIdx.x * 32 + rowL;
            float mu = s_sum[rowL] * (1.f / Cc);
            float var = s_sq[rowL] * (1.f / Cc) - mu * mu;
            float rstd = rsqrtf(var + EPSl);
            float o = (acc[ct][r] - mu) * rstd * gg + bb2;
            Y[(size_t)m * Cc + col]  = o;
            Yh[(size_t)m * Cc + col] = (_Float16)o;
        }
    }
}

// =====================================================================
// 4) small per-layer node GEMMs: ai,aj,tn,nf (each (B,N,C), K=256)
// =====================================================================
__global__ void k_small_mats(const float* __restrict__ nodes,
                             const float* __restrict__ Wb_l,
                             const float* __restrict__ Wfp_l,
                             const float* __restrict__ Wf_l,
                             float* __restrict__ ai, float* __restrict__ aj,
                             float* __restrict__ tn, float* __restrict__ nf) {
    __shared__ float sn[Cc];
    int row = blockIdx.x, which = blockIdx.y, col = threadIdx.x;
    sn[col] = nodes[row * Cc + col];
    __syncthreads();
    const float* w;
    float* out;
    if (which == 0)      { w = Wb_l + (size_t)col * (3 * Cc);            out = ai; }
    else if (which == 1) { w = Wb_l + (size_t)col * (3 * Cc) + 2 * Cc;   out = aj; }
    else if (which == 2) { w = Wfp_l + (size_t)col * Cc;                 out = tn; }
    else                 { w = Wf_l + (size_t)col * Cc;                  out = nf; }
    float acc = 0.f;
    #pragma unroll 8
    for (int k = 0; k < Cc; ++k) acc += sn[k] * w[k];
    out[row * Cc + col] = acc;
}

// s[row] = tn[row,:] . ub   |   v[n] = sum_c Wbp[c,n]*ub[c]
__global__ void k_svec(const float* __restrict__ tn,
                       const float* __restrict__ Wbp_l,
                       const float* __restrict__ ub_l,
                       float* __restrict__ s, float* __restrict__ v) {
    int t = threadIdx.x;
    if (blockIdx.x == 0) {
        const float* r = tn + t * Cc;
        float acc = 0.f;
        for (int k = 0; k < Cc; ++k) acc += r[k] * ub_l[k];
        s[t] = acc;
    } else {
        float acc = 0.f;
        for (int c = 0; c < Cc; ++c) acc += Wbp_l[c * Cc + t] * ub_l[c];
        v[t] = acc;
    }
}

// =====================================================================
// 5) scores + masked softmax over j (folded matvec form)
// =====================================================================
__global__ void k_score_softmax(const float* __restrict__ NE,
                                const float* __restrict__ v,
                                const float* __restrict__ s,
                                const float* __restrict__ mask,
                                float* __restrict__ attn) {
    __shared__ float red[Nn];
    int bi = blockIdx.x;
    int b = bi >> 7, i = bi & 127;
    int j = threadIdx.x;
    const float* row = NE + (size_t)(((b << 7) + j) * Nn + i) * Cc;
    float acc = 0.f;
    #pragma unroll 8
    for (int c = 0; c < Cc; ++c) acc += row[c] * v[c];
    float r = lrelu(acc + s[(b << 7) + i] + s[(b << 7) + j]);
    if (mask[((b << 7) + i) * Nn + j] <= 0.f) r = -INFINITY;

    red[j] = r; __syncthreads();
    for (int off = 64; off > 0; off >>= 1) {
        if (j < off) red[j] = fmaxf(red[j], red[j + off]);
        __syncthreads();
    }
    float mx = red[0]; __syncthreads();
    float e = (r == -INFINITY) ? 0.f : __expf(r - mx);
    red[j] = e; __syncthreads();
    for (int off = 64; off > 0; off >>= 1) {
        if (j < off) red[j] += red[j + off];
        __syncthreads();
    }
    float sum = red[0];
    attn[(size_t)bi * Nn + j] = (sum > 0.f) ? e / sum : 0.f;
}

// =====================================================================
// 6) agg = attn @ nf ; nodes = LN(lrelu(agg)+nodes)  (in-place per row)
// =====================================================================
__global__ void k_agg_ln(const float* __restrict__ attn,
                         const float* __restrict__ nf,
                         float* __restrict__ nodes,
                         const float* __restrict__ g,
                         const float* __restrict__ bvec) {
    __shared__ float sat[Nn];
    __shared__ float red[Cc];
    int bi = blockIdx.x;
    int b = bi >> 7;
    int c = threadIdx.x;
    if (c < Nn) sat[c] = attn[(size_t)bi * Nn + c];
    __syncthreads();
    float acc = 0.f;
    #pragma unroll 4
    for (int j = 0; j < Nn; ++j) acc += sat[j] * nf[((b << 7) + j) * Cc + c];
    float x = lrelu(acc) + nodes[bi * Cc + c];

    red[c] = x; __syncthreads();
    for (int off = 128; off > 0; off >>= 1) {
        if (c < off) red[c] += red[c + off];
        __syncthreads();
    }
    float mu = red[0] * (1.f / Cc); __syncthreads();
    float d = x - mu;
    red[c] = d * d; __syncthreads();
    for (int off = 128; off > 0; off >>= 1) {
        if (c < off) red[c] += red[c + off];
        __syncthreads();
    }
    float rstd = rsqrtf(red[0] * (1.f / Cc) + EPSl);
    __syncthreads();
    nodes[bi * Cc + c] = d * rstd * g[c] + bvec[c];
}

// =====================================================================
// 7) outputs
// =====================================================================
__global__ void k_node_out(const float* __restrict__ nodes,
                           const float* __restrict__ ncW,
                           const float* __restrict__ ncb,
                           float* __restrict__ out_nodes,
                           float* __restrict__ out_logits) {
    __shared__ float sn[Cc];
    int row = blockIdx.x, t = threadIdx.x;      // 128 threads
    sn[t] = nodes[row * Cc + t];
    sn[t + 128] = nodes[row * Cc + t + 128];
    __syncthreads();
    out_nodes[row * Cc + t] = sn[t];
    out_nodes[row * Cc + t + 128] = sn[t + 128];
    if (t < KN) {
        const float* w = ncW + t * Cc;
        float acc = ncb[t];
        for (int k = 0; k < Cc; ++k) acc += sn[k] * w[k];
        out_logits[row * KN + t] = acc;
    }
}

__global__ void k_edge_logits(const float* __restrict__ E,
                              const float* __restrict__ ecW,
                              const float* __restrict__ ecb,
                              float* __restrict__ out) {
    int t = threadIdx.x;
    int row = blockIdx.x * 32 + (t >> 3);
    int o = t & 7;
    if (o >= KE) return;
    const float* r = E + (size_t)row * Cc;
    const float* w = ecW + o * Cc;
    float acc = ecb[o];
    #pragma unroll 8
    for (int k = 0; k < Cc; ++k) acc += r[k] * w[k];
    out[row * KE + o] = acc;
}

// =====================================================================
extern "C" void kernel_launch(void* const* d_in, const int* in_sizes, int n_in,
                              void* d_out, int out_size, void* d_ws, size_t ws_size,
                              hipStream_t stream) {
    const float* in_nodes = (const float*)d_in[0];
    const float* in_edges = (const float*)d_in[1];
    const float* mask     = (const float*)d_in[2];
    const float* pnW      = (const float*)d_in[3];
    const float* pnb      = (const float*)d_in[4];
    const float* peW      = (const float*)d_in[5];
    const float* peb      = (const float*)d_in[6];
    const float* Wb       = (const float*)d_in[7];
    const float* Wbp      = (const float*)d_in[8];
    const float* Wfp      = (const float*)d_in[9];
    const float* ub       = (const float*)d_in[10];
    const float* Wf       = (const float*)d_in[11];
    const float* eln_g    = (const float*)d_in[12];
    const float* eln_b    = (const float*)d_in[13];
    const float* nln_g    = (const float*)d_in[14];
    const float* nln_b    = (const float*)d_in[15];
    const float* ncW      = (const float*)d_in[16];
    const float* ncb      = (const float*)d_in[17];
    const float* ecW      = (const float*)d_in[18];
    const float* ecb      = (const float*)d_in[19];

    float* ws = (float*)d_ws;
    const size_t EM = (size_t)Mrows * Cc;        // 8,388,608
    float* edgesA = ws;
    float* edgesB = ws + EM;
    float* nodes  = ws + 2 * EM;
    float* ai     = nodes + 65536;
    float* aj     = ai + 65536;
    float* tn     = aj + 65536;
    float* nf     = tn + 65536;
    float* sbuf   = nf + 65536;
    float* vbuf   = sbuf + 256;
    float* attn   = vbuf + 256;
    float* fend   = attn + (size_t)Mrows;        // end of fp32 region

    _Float16* edgesAh = (_Float16*)fend;                       // M*C halfs
    _Float16* edgesBh = edgesAh + EM;
    _Float16* rawh    = edgesBh + EM;                          // M*64 halfs
    _Float16* W2h     = rawh + (size_t)Mrows * EDGE_IN;        // 256*256 halfs
    _Float16* peWh    = W2h + (size_t)Cc * Cc;                 // 256*64 halfs

    float* out_f        = (float*)d_out;
    float* out_nodes    = out_f;
    float* out_edges    = out_f + 65536;
    float* out_nlogits  = out_f + 65536 + EM;
    float* out_elogits  = out_f + 65536 + EM + (size_t)Bx * Nn * KN;

    // operand conversions (one-shot, tiny vs. GEMM traffic)
    k_f32_to_f16<<<(Mrows * EDGE_IN) / 256, 256, 0, stream>>>(
        in_edges, rawh, Mrows * EDGE_IN);
    k_w_to_f16<<<Cc, 256, 0, stream>>>(peW, EDGE_IN, EDGE_IN, peWh);

    // projections
    k_node_proj<<<Bx * Nn, Cc, 0, stream>>>(in_nodes, pnW, pnb, nodes);
    k_edge_proj_wmma<<<Mrows / 32, 256, 0, stream>>>(rawh, peWh, peb, edgesA, edgesAh);

    float*    ebuf[2]  = {edgesA, edgesB};
    _Float16* ehbuf[2] = {edgesAh, edgesBh};
    int cur = 0;
    for (int l = 0; l < Ll; ++l) {
        const float* Wb_l  = Wb  + (size_t)l * Cc * 3 * Cc;
        const float* Wbp_l = Wbp + (size_t)l * Cc * Cc;
        const float* Wfp_l = Wfp + (size_t)l * Cc * Cc;
        const float* Wf_l  = Wf  + (size_t)l * Cc * Cc;
        const float* ub_l  = ub  + (size_t)l * Cc;

        float*    ein  = ebuf[cur];
        _Float16* einh = ehbuf[cur];
        float*    eout = (l == Ll - 1) ? out_edges : ebuf[1 - cur];
        _Float16* eouth = ehbuf[1 - cur];

        k_w_to_f16<<<Cc, 256, 0, stream>>>(Wb_l + Cc, 3 * Cc, Cc, W2h);   // Wb2 slice
        k_small_mats<<<dim3(Bx * Nn, 4), Cc, 0, stream>>>(
            nodes, Wb_l, Wfp_l, Wf_l, ai, aj, tn, nf);
        k_svec<<<2, Cc, 0, stream>>>(tn, Wbp_l, ub_l, sbuf, vbuf);
        k_edge_layer_wmma<<<Mrows / 32, 256, 0, stream>>>(
            ein, einh, W2h, ai, aj, mask,
            eln_g + l * Cc, eln_b + l * Cc, eout, eouth);
        k_score_softmax<<<Bx * Nn, Nn, 0, stream>>>(eout, vbuf, sbuf, mask, attn);
        k_agg_ln<<<Bx * Nn, Cc, 0, stream>>>(attn, nf, nodes,
                                             nln_g + l * Cc, nln_b + l * Cc);
        cur = 1 - cur;
    }

    k_node_out<<<Bx * Nn, 128, 0, stream>>>(nodes, ncW, ncb, out_nodes, out_nlogits);
    k_edge_logits<<<Mrows / 32, 256, 0, stream>>>(out_edges, ecW, ecb, out_elogits);
}
